// Generator_51049981280754
// MI455X (gfx1250) — matrix operand order repaired and checked by reading, not compiled, not run on hardware
//
#include <hip/hip_runtime.h>

#define Bb   2
#define Ss   1024
#define Dd   1024
#define Hh   16
#define Ll   8
#define FFf  4096
#define Vv   4096
#define DHd  64
#define BSn  (Bb*Ss)

typedef unsigned short ushort_t;
typedef __attribute__((ext_vector_type(16))) __bf16 v16bf;
typedef __attribute__((ext_vector_type(8)))  __bf16 v8bf;
typedef __attribute__((ext_vector_type(8)))  float  v8f;

__device__ __forceinline__ ushort_t f2bf(float f) {
  unsigned int u = __float_as_uint(f);
  u += 0x7FFFu + ((u >> 16) & 1u);      // round-to-nearest-even
  return (ushort_t)(u >> 16);
}
__device__ __forceinline__ float gelu_tanh(float x) {
  float u = 0.7978845608028654f * (x + 0.044715f * x * x * x);
  return 0.5f * x * (1.0f + tanhf(u));
}

// ---- CDNA5 async global->LDS copy (ASYNCcnt path), 16 bytes per lane ----
__device__ __forceinline__ void async_copy_b128(unsigned int lds_byte_off,
                                                const void* gaddr) {
  // ISA 15.18.3 opcode 98: GLOBAL_LOAD_ASYNC_TO_LDS_B128 vdst(LDS addr), vaddr, off
  asm volatile("global_load_async_to_lds_b128 %0, %1, off"
               :: "v"(lds_byte_off), "v"(gaddr)
               : "memory");
}
__device__ __forceinline__ void wait_async_zero() {
#if __has_builtin(__builtin_amdgcn_s_wait_asynccnt)
  __builtin_amdgcn_s_wait_asynccnt(0);
#else
  asm volatile("s_wait_asynccnt 0" ::: "memory");
#endif
}

// -------- RoPE tables: cos/sin (S x 32) --------
__global__ void rope_tables_kernel(float* __restrict__ ct, float* __restrict__ st) {
  int idx = blockIdx.x * 256 + threadIdx.x;      // S*32 = 32768
  int s = idx >> 5, i = idx & 31;
  float invf = expf(-((float)(2 * i) / 64.0f) * 9.210340371976184f); // ln(10000)
  float a = (float)s * invf;
  ct[idx] = cosf(a);
  st[idx] = sinf(a);
}

// -------- Embedding + positional, fp32 + bf16 copies --------
__global__ __launch_bounds__(256)
void embed_kernel(const int* __restrict__ inp, const float* __restrict__ emb,
                  const float* __restrict__ pos, float* __restrict__ x,
                  ushort_t* __restrict__ xh) {
  int row = blockIdx.x;                 // 0..BS-1
  int sidx = row & (Ss - 1);
  int tok = inp[row];
  int d = threadIdx.x * 4;
  float4 e = *(const float4*)(emb + (size_t)tok * Dd + d);
  float4 p = *(const float4*)(pos + (size_t)sidx * Dd + d);
  float o0 = e.x + p.x, o1 = e.y + p.y, o2 = e.z + p.z, o3 = e.w + p.w;
  *(float4*)(x + (size_t)row * Dd + d) = make_float4(o0, o1, o2, o3);
  uint2 pk;
  pk.x = (unsigned)f2bf(o0) | ((unsigned)f2bf(o1) << 16);
  pk.y = (unsigned)f2bf(o2) | ((unsigned)f2bf(o3) << 16);
  *(uint2*)(xh + (size_t)row * Dd + d) = pk;
}

// -------- Transposing fp32 -> bf16 weight convert: W[K][N] -> Wt[N][K] --------
__global__ void tcvt_kernel(const float* __restrict__ W, ushort_t* __restrict__ Wt,
                            int K, int N) {
  __shared__ float tile[32][33];
  int n0 = blockIdx.x * 32, k0 = blockIdx.y * 32;
  int tx = threadIdx.x, ty = threadIdx.y;
  for (int i = ty; i < 32; i += 8)
    tile[i][tx] = W[(size_t)(k0 + i) * N + n0 + tx];
  __syncthreads();
  for (int i = ty; i < 32; i += 8)
    Wt[(size_t)(n0 + i) * K + k0 + tx] = f2bf(tile[tx][i]);
}

// -------- bf16 WMMA GEMM: C[M][N] = A[M][K] * Bt[N][K]^T (+bias,+res,GELU) ----
// Block tile 128x128, K tile 64, 256 threads = 8 waves (4 M x 2 N),
// wave tile 32x64 = 2x4 16x16 WMMA accumulators.
// Double-buffered LDS; next tile is fetched with global_load_async_to_lds_b128
// (ASYNCcnt) while the current tile feeds the WMMAs.
__device__ __forceinline__ v16bf make_frag(const ushort_t* p0, const ushort_t* p1) {
  v8bf lo = *(const v8bf*)p0;
  v8bf hi = *(const v8bf*)p1;
  v16bf r;
#pragma unroll
  for (int i = 0; i < 8; ++i) { r[i] = lo[i]; r[i + 8] = hi[i]; }
  return r;
}

__global__ __launch_bounds__(256)
void gemm_bf16_kernel(const ushort_t* __restrict__ A, const ushort_t* __restrict__ Bt,
                      const float* __restrict__ bias, const float* __restrict__ res,
                      float* __restrict__ Cf, ushort_t* __restrict__ Ch,
                      int M, int N, int K, int act) {
  constexpr int BM = 128, BN = 128, BK = 64, LD = 72; // 144B row pitch, 16B aligned
  __shared__ alignas(16) ushort_t As[2][BM * LD];
  __shared__ alignas(16) ushort_t Bs[2][BN * LD];
  constexpr unsigned int BUFB = (unsigned int)(BM * LD * 2); // bytes per buffer
  int tid = threadIdx.x;
  int lane = tid & 31, w = tid >> 5;
  int wm = w & 3, wn = w >> 2;
  int m0 = blockIdx.y * BM, n0 = blockIdx.x * BN;
  int half = lane >> 4, r16 = lane & 15;

  v8f acc[2][4];
#pragma unroll
  for (int a = 0; a < 2; ++a)
#pragma unroll
    for (int b = 0; b < 4; ++b)
#pragma unroll
      for (int i = 0; i < 8; ++i) acc[a][b][i] = 0.0f;

  int lrow = tid >> 1, loff = (tid & 1) * 32;   // 64 halfs per row, 2 threads/row
  const ushort_t* ag = A + (size_t)(m0 + lrow) * K + loff;
  const ushort_t* bg = Bt + (size_t)(n0 + lrow) * K + loff;
  // LDS byte offsets for this lane's staging slots (flat-pointer low 32 bits)
  unsigned int aoff0 = (unsigned int)(uintptr_t)&As[0][lrow * LD + loff];
  unsigned int boff0 = (unsigned int)(uintptr_t)&Bs[0][lrow * LD + loff];

  // Prologue: stage tile 0 into buffer 0 via async DMA
#pragma unroll
  for (int i = 0; i < 4; ++i) {
    async_copy_b128(aoff0 + i * 16, ag + i * 8);
    async_copy_b128(boff0 + i * 16, bg + i * 8);
  }
  wait_async_zero();
  __syncthreads();

  int nk = K / BK;
  for (int t = 0; t < nk; ++t) {
    int cur = t & 1;
    if (t + 1 < nk) {                     // prefetch next tile into other buffer
      int kn = (t + 1) * BK;
      unsigned int ao = aoff0 + (unsigned int)(cur ^ 1) * BUFB;
      unsigned int bo = boff0 + (unsigned int)(cur ^ 1) * BUFB;
#pragma unroll
      for (int i = 0; i < 4; ++i) {
        async_copy_b128(ao + i * 16, ag + kn + i * 8);
        async_copy_b128(bo + i * 16, bg + kn + i * 8);
      }
    }
    const ushort_t* Abuf = As[cur];
    const ushort_t* Bbuf = Bs[cur];
#pragma unroll
    for (int ks = 0; ks < BK; ks += 32) {
      v16bf af[2], bfr[4];
#pragma unroll
      for (int sm = 0; sm < 2; ++sm) {
        int row = wm * 32 + sm * 16 + r16;
        const ushort_t* base = Abuf + row * LD + ks + half * 8;   // kbase = half*8
        af[sm] = make_frag(base, base + 16);                      // K, K+16 chunks
      }
#pragma unroll
      for (int sn = 0; sn < 4; ++sn) {
        int col = wn * 64 + sn * 16 + r16;
        const ushort_t* base = Bbuf + col * LD + ks + half * 8;   // kbase = half*8
        bfr[sn] = make_frag(base, base + 16);
      }
#pragma unroll
      for (int sm = 0; sm < 2; ++sm)
#pragma unroll
        for (int sn = 0; sn < 4; ++sn)
          acc[sm][sn] = __builtin_amdgcn_wmma_f32_16x16x32_bf16(
              false, af[sm], false, bfr[sn], (short)0, acc[sm][sn], false, false);
    }
    wait_async_zero();   // own async loads for next tile done
    __syncthreads();     // everyone done computing cur + loads visible
  }

#pragma unroll
  for (int sm = 0; sm < 2; ++sm)
#pragma unroll
    for (int sn = 0; sn < 4; ++sn)
#pragma unroll
      for (int i = 0; i < 8; ++i) {
        int row = m0 + wm * 32 + sm * 16 + ((lane < 16) ? i : i + 8);
        int col = n0 + wn * 64 + sn * 16 + r16;
        float v = acc[sm][sn][i] + bias[col];
        if (res) v += res[(size_t)row * N + col];
        if (act) v = gelu_tanh(v);
        if (Cf) Cf[(size_t)row * N + col] = v;
        if (Ch) Ch[(size_t)row * N + col] = f2bf(v);
      }
}

// -------- Linear attention scan (RoPE + ELU+1 features + inclusive cumsum) ----
// One block per (b,h). kv state [64][64] in registers: thread (e,g) owns
// kv[d0..d0+15][e], d0 = g*16. Output written directly as bf16.
__global__ __launch_bounds__(256)
void linattn_kernel(const float* __restrict__ q, const float* __restrict__ k,
                    const float* __restrict__ v, const int* __restrict__ mask,
                    const float* __restrict__ ct, const float* __restrict__ st,
                    ushort_t* __restrict__ ah) {
  int bh = blockIdx.x;
  int b = bh >> 4, h = bh & 15;
  int t = threadIdx.x;
  int e = t & 63, g = t >> 6, d0 = g << 4;
  __shared__ float qfs[64], kfs[64], vvs[64], zps[64], s8[8], part[256];
  float kv[16];
#pragma unroll
  for (int i = 0; i < 16; ++i) kv[i] = 0.0f;
  float ksum = 0.0f;                 // valid for t<64 (dim index t)
  int j = t;
  int pj = (j < 32) ? j + 32 : j - 32;
  float sgn = (j < 32) ? -1.0f : 1.0f;

  for (int s = 0; s < Ss; ++s) {
    if (t < 64) {
      size_t base = ((size_t)(b * Ss + s)) * Dd + h * DHd;
      float c = ct[s * 32 + (j & 31)];
      float sn = st[s * 32 + (j & 31)];
      float qv = q[base + j], qp = q[base + pj];
      float kl = k[base + j], kp = k[base + pj];
      float qr = qv * c + sgn * qp * sn;
      float kr = kl * c + sgn * kp * sn;
      float qf = (qr > 0.0f) ? qr + 1.0f : expf(qr);   // elu+1
      float kf = ((kr > 0.0f) ? kr + 1.0f : expf(kr)) * (float)mask[b * Ss + s];
      ksum += kf;                                      // inclusive cumsum
      qfs[j] = qf; kfs[j] = kf; vvs[j] = v[base + j]; zps[j] = qf * ksum;
    }
    __syncthreads();
    {
      float vve = vvs[e];
      float np = 0.0f;
#pragma unroll
      for (int i = 0; i < 16; ++i) {
        kv[i] = fmaf(kfs[d0 + i], vve, kv[i]);         // inclusive state update
        np = fmaf(qfs[d0 + i], kv[i], np);
      }
      part[(g << 6) | e] = np;
      if (t < 8) {
        float ssum = 0.0f;
#pragma unroll
        for (int i = 0; i < 8; ++i) ssum += zps[(t << 3) + i];
        s8[t] = ssum;
      }
    }
    __syncthreads();
    if (t < 64) {
      float z = s8[0] + s8[1] + s8[2] + s8[3] + s8[4] + s8[5] + s8[6] + s8[7];
      float num = part[e] + part[64 + e] + part[128 + e] + part[192 + e];
      float a = num / (z + 1e-6f);
      ah[((size_t)(b * Ss + s)) * Dd + h * DHd + e] = f2bf(a);
    }
    __syncthreads();
  }
}

// -------- Conditional LayerNorm (per-token), fp32 + bf16 outputs --------
__global__ __launch_bounds__(256)
void cln_kernel(const float* __restrict__ in, const float* __restrict__ g,
                const float* __restrict__ bta, const int* __restrict__ cond,
                float* __restrict__ outf, ushort_t* __restrict__ outh) {
  __shared__ float r1[256], r2[256];
  int row = blockIdx.x;
  int b = row >> 10;                  // / S
  int t = threadIdx.x;
  int d = t * 4;
  float4 x4 = *(const float4*)(in + (size_t)row * Dd + d);
  r1[t] = x4.x + x4.y + x4.z + x4.w;
  r2[t] = x4.x * x4.x + x4.y * x4.y + x4.z * x4.z + x4.w * x4.w;
  __syncthreads();
  for (int off = 128; off > 0; off >>= 1) {
    if (t < off) { r1[t] += r1[t + off]; r2[t] += r2[t + off]; }
    __syncthreads();
  }
  float mu = r1[0] * (1.0f / Dd);
  float var = r2[0] * (1.0f / Dd) - mu * mu;
  float rs = rsqrtf(var + 1e-5f);
  int c = cond[b];
  float4 gg = *(const float4*)(g + (size_t)c * Dd + d);
  float4 bb = *(const float4*)(bta + (size_t)c * Dd + d);
  float o0 = (x4.x - mu) * rs * gg.x + bb.x;
  float o1 = (x4.y - mu) * rs * gg.y + bb.y;
  float o2 = (x4.z - mu) * rs * gg.z + bb.z;
  float o3 = (x4.w - mu) * rs * gg.w + bb.w;
  *(float4*)(outf + (size_t)row * Dd + d) = make_float4(o0, o1, o2, o3);
  uint2 pk;
  pk.x = (unsigned)f2bf(o0) | ((unsigned)f2bf(o1) << 16);
  pk.y = (unsigned)f2bf(o2) | ((unsigned)f2bf(o3) << 16);
  *(uint2*)(outh + (size_t)row * Dd + d) = pk;
}

// -------- Gumbel softmax + straight-through one-hot: (y1 - y) + y --------
__global__ __launch_bounds__(256)
void gumbel_kernel(const float* __restrict__ logits, const float* __restrict__ u,
                   const float* __restrict__ inv_temp, float* __restrict__ og) {
  __shared__ float rv[256];
  __shared__ int ri[256];
  int row = blockIdx.x;
  int t = threadIdx.x;
  const float* o = logits + (size_t)row * Vv;
  const float* uu = u + (size_t)row * Vv;
  float it = inv_temp[0];
  float tv[16];
  float mx = -3.4e38f;
  int mi = 0;
#pragma unroll
  for (int i = 0; i < 16; ++i) {
    int j = t * 16 + i;
    float gg = -logf(-logf(uu[j] + 1e-9f) + 1e-9f);
    float tt = (o[j] + gg) * it;
    tv[i] = tt;
    if (tt > mx) { mx = tt; mi = j; }          // first occurrence within thread
  }
  rv[t] = mx; ri[t] = mi;
  __syncthreads();
  for (int off = 128; off > 0; off >>= 1) {
    if (t < off) {
      float av = rv[t], bv = rv[t + off];
      int ai = ri[t], bi = ri[t + off];
      if (bv > av || (bv == av && bi < ai)) { rv[t] = bv; ri[t] = bi; }
    }
    __syncthreads();
  }
  float MX = rv[0];
  int MI = ri[0];
  __syncthreads();
  float lsum = 0.0f;
#pragma unroll
  for (int i = 0; i < 16; ++i) lsum += expf(tv[i] - MX);
  rv[t] = lsum;
  __syncthreads();
  for (int off = 128; off > 0; off >>= 1) {
    if (t < off) rv[t] += rv[t + off];
    __syncthreads();
  }
  float inv = 1.0f / rv[0];
#pragma unroll
  for (int i = 0; i < 16; ++i) {
    int j = t * 16 + i;
    float y = expf(tv[i] - MX) * inv;
    float y1 = (j == MI) ? 1.0f : 0.0f;
    og[(size_t)row * Vv + j] = (y1 - y) + y;   // same expression as reference
  }
}

extern "C" void kernel_launch(void* const* d_in, const int* in_sizes, int n_in,
                              void* d_out, int out_size, void* d_ws, size_t ws_size,
                              hipStream_t stream) {
  (void)in_sizes; (void)n_in; (void)out_size; (void)ws_size;
  const int*   inputs    = (const int*)  d_in[0];
  const int*   cond      = (const int*)  d_in[1];
  const int*   mask      = (const int*)  d_in[2];
  const float* gumbel_u  = (const float*)d_in[3];
  const float* inv_temp  = (const float*)d_in[4];
  const float* embedding = (const float*)d_in[5];
  const float* pos_emb   = (const float*)d_in[6];
  const float* Wq = (const float*)d_in[7];
  const float* bq = (const float*)d_in[8];
  const float* Wk = (const float*)d_in[9];
  const float* bk = (const float*)d_in[10];
  const float* Wv = (const float*)d_in[11];
  const float* bv = (const float*)d_in[12];
  const float* Wo = (const float*)d_in[13];
  const float* bo = (const float*)d_in[14];
  const float* ln1_g = (const float*)d_in[15];
  const float* ln1_b = (const float*)d_in[16];
  const float* W1 = (const float*)d_in[17];
  const float* b1 = (const float*)d_in[18];
  const float* W2 = (const float*)d_in[19];
  const float* b2 = (const float*)d_in[20];
  const float* ln2_g = (const float*)d_in[21];
  const float* ln2_b = (const float*)d_in[22];
  const float* normf_g = (const float*)d_in[23];
  const float* normf_b = (const float*)d_in[24];
  const float* Wout = (const float*)d_in[25];
  const float* bout = (const float*)d_in[26];

  // Workspace carve-up (~72.5 MB total)
  char* wsp = (char*)d_ws;
  size_t off = 0;
  auto alloc = [&](size_t bytes) -> void* {
    void* p = wsp + off;
    off += (bytes + 255) & ~(size_t)255;
    return p;
  };
  float*    x   = (float*)   alloc((size_t)BSn * Dd * 4);
  ushort_t* xh  = (ushort_t*)alloc((size_t)BSn * Dd * 2);
  float*    tmp = (float*)   alloc((size_t)BSn * Dd * 4);
  float*    qb  = (float*)   alloc((size_t)BSn * Dd * 4);
  float*    kb  = (float*)   alloc((size_t)BSn * Dd * 4);
  float*    vb  = (float*)   alloc((size_t)BSn * Dd * 4);
  ushort_t* ahb = (ushort_t*)alloc((size_t)BSn * Dd * 2);
  ushort_t* yh  = (ushort_t*)alloc((size_t)BSn * FFf * 2);
  ushort_t* wst = (ushort_t*)alloc((size_t)Dd * FFf * 2);   // weight staging (bf16, transposed)
  float*    ct  = (float*)   alloc((size_t)Ss * 32 * 4);
  float*    st  = (float*)   alloc((size_t)Ss * 32 * 4);

  rope_tables_kernel<<<(Ss * 32) / 256, 256, 0, stream>>>(ct, st);
  embed_kernel<<<BSn, 256, 0, stream>>>(inputs, embedding, pos_emb, x, xh);

  dim3 tb(32, 8);
  for (int l = 0; l < Ll; ++l) {
    tcvt_kernel<<<dim3(Dd / 32, Dd / 32), tb, 0, stream>>>(Wq + (size_t)l * Dd * Dd, wst, Dd, Dd);
    gemm_bf16_kernel<<<dim3(Dd / 128, BSn / 128), 256, 0, stream>>>(
        xh, wst, bq + l * Dd, nullptr, qb, nullptr, BSn, Dd, Dd, 0);
    tcvt_kernel<<<dim3(Dd / 32, Dd / 32), tb, 0, stream>>>(Wk + (size_t)l * Dd * Dd, wst, Dd, Dd);
    gemm_bf16_kernel<<<dim3(Dd / 128, BSn / 128), 256, 0, stream>>>(
        xh, wst, bk + l * Dd, nullptr, kb, nullptr, BSn, Dd, Dd, 0);
    tcvt_kernel<<<dim3(Dd / 32, Dd / 32), tb, 0, stream>>>(Wv + (size_t)l * Dd * Dd, wst, Dd, Dd);
    gemm_bf16_kernel<<<dim3(Dd / 128, BSn / 128), 256, 0, stream>>>(
        xh, wst, bv + l * Dd, nullptr, vb, nullptr, BSn, Dd, Dd, 0);

    linattn_kernel<<<Bb * Hh, 256, 0, stream>>>(qb, kb, vb, mask, ct, st, ahb);

    tcvt_kernel<<<dim3(Dd / 32, Dd / 32), tb, 0, stream>>>(Wo + (size_t)l * Dd * Dd, wst, Dd, Dd);
    gemm_bf16_kernel<<<dim3(Dd / 128, BSn / 128), 256, 0, stream>>>(
        ahb, wst, bo + l * Dd, x, tmp, nullptr, BSn, Dd, Dd, 0);
    cln_kernel<<<BSn, 256, 0, stream>>>(tmp, ln1_g + (size_t)l * 4 * Dd,
                                        ln1_b + (size_t)l * 4 * Dd, cond, x, xh);

    tcvt_kernel<<<dim3(FFf / 32, Dd / 32), tb, 0, stream>>>(W1 + (size_t)l * Dd * FFf, wst, Dd, FFf);
    gemm_bf16_kernel<<<dim3(FFf / 128, BSn / 128), 256, 0, stream>>>(
        xh, wst, b1 + l * FFf, nullptr, nullptr, yh, BSn, FFf, Dd, 1);   // fused GELU -> bf16
    tcvt_kernel<<<dim3(Dd / 32, FFf / 32), tb, 0, stream>>>(W2 + (size_t)l * FFf * Dd, wst, FFf, Dd);
    gemm_bf16_kernel<<<dim3(Dd / 128, BSn / 128), 256, 0, stream>>>(
        yh, wst, b2 + l * Dd, x, tmp, nullptr, BSn, Dd, FFf, 0);
    cln_kernel<<<BSn, 256, 0, stream>>>(tmp, ln2_g + (size_t)l * 4 * Dd,
                                        ln2_b + (size_t)l * 4 * Dd, cond, x, xh);
  }

  cln_kernel<<<BSn, 256, 0, stream>>>(x, normf_g, normf_b, cond, x, xh);
  tcvt_kernel<<<dim3(Vv / 32, Dd / 32), tb, 0, stream>>>(Wout, wst, Dd, Vv);
  gemm_bf16_kernel<<<dim3(Vv / 128, BSn / 128), 256, 0, stream>>>(
      xh, wst, bout, nullptr, (float*)d_out, nullptr, BSn, Vv, Dd, 0);
  gumbel_kernel<<<BSn, 256, 0, stream>>>((float*)d_out, gumbel_u, inv_temp,
                                         (float*)d_out + (size_t)BSn * Vv);
}
